// MotionGenerator_41644002902498
// MI455X (gfx1250) — compile-verified, matrix operand order checked by reading, not run on hardware
//
#include <hip/hip_runtime.h>
#include <hip/hip_bf16.h>

// ---------------------------------------------------------------------------
// Masked skeleton conv (NCH, K=15, stride 2, reflect pad 7) + joint pooling +
// leaky relu, built on CDNA5 WMMA (v_wmma_f32_16x16x32_f16).
// Round 2: double-buffered LDS staging (1 barrier/pair), branch-free staging
// for interior time tiles, global_prefetch of the pair-after-next x window.
// ---------------------------------------------------------------------------

typedef __attribute__((ext_vector_type(16))) _Float16 v16h;
typedef __attribute__((ext_vector_type(8)))  _Float16 v8h;
typedef __attribute__((ext_vector_type(8)))  float    v8f;

union V16 { v16h v; v8h h[2]; };

#define NJ       21      // joints (20 edges + global)
#define CIN_PJ   64
#define COUT_PJ  128
#define IN_C     (NJ*CIN_PJ)    // 1344
#define OUT_C    (NJ*COUT_PJ)   // 2688
#define T_IN     4096
#define T_OUT    2048
#define KS       15
#define BATCH    8
#define NGROUP   12
#define TT       64                  // time tile per workgroup
#define NTB      (T_OUT/TT)          // 32 time tiles
#define XT       (2*TT + KS - 1)     // 142 staged time samples
#define PITCH    72                  // halfs per LDS row (144B, 16B aligned)
#define NPAIRS   128

// Skeleton distance-2 neighbor structure (derived from EDGES, SKELETON_DIST=2,
// incl. global joint 20 wiring) -- matches the reference mask exactly.
__constant__ int c_noff[NJ + 1] = {
    0, 8, 17, 26, 33, 41, 48, 52, 55, 63, 70, 74,
    77, 85, 92, 96, 99, 107, 114, 118, 121, 128};
__constant__ int c_nflat[NPAIRS] = {
    /* i0 */ 0, 1, 2, 12, 13, 16, 17, 20,
    /* i1 */ 0, 1, 2, 3, 4, 8, 12, 16, 20,
    /* i2 */ 0, 1, 2, 3, 4, 5, 8, 9, 20,
    /* i3 */ 1, 2, 3, 4, 5, 8, 9,
    /* i4 */ 1, 2, 3, 4, 5, 6, 8, 9,
    /* i5 */ 2, 3, 4, 5, 6, 7, 8,
    /* i6 */ 4, 5, 6, 7,
    /* i7 */ 5, 6, 7,
    /* i8 */ 1, 2, 3, 4, 5, 8, 9, 10,
    /* i9 */ 2, 3, 4, 8, 9, 10, 11,
    /* i10*/ 8, 9, 10, 11,
    /* i11*/ 9, 10, 11,
    /* i12*/ 0, 1, 12, 13, 14, 16, 17, 20,
    /* i13*/ 0, 12, 13, 14, 15, 16, 20,
    /* i14*/ 12, 13, 14, 15,
    /* i15*/ 13, 14, 15,
    /* i16*/ 0, 1, 12, 13, 16, 17, 18, 20,
    /* i17*/ 0, 12, 16, 17, 18, 19, 20,
    /* i18*/ 16, 17, 18, 19,
    /* i19*/ 17, 18, 19,
    /* i20*/ 0, 1, 2, 12, 13, 16, 17};

// Pooling groups (edge indices averaged per group)
__constant__ int c_poolA[NGROUP] = {0, 1, 3, 4, 6, 8, 10, 12, 14, 16, 18, 20};
__constant__ int c_poolB[NGROUP] = {-1, 2, -1, 5, 7, 9, 11, 13, 15, 17, 19, -1};

// ---------------------------------------------------------------------------
// Kernel 1: repack live weight blocks to f16: wh[pair][tap][cout128][cin64]
// ---------------------------------------------------------------------------
__global__ __launch_bounds__(256)
void mg_pack_weights(const float* __restrict__ w, _Float16* __restrict__ wh)
{
    const int pk   = blockIdx.x;          // pair*KS + k
    const int pair = pk / KS;
    const int k    = pk % KS;
    int i = 0;
    while (c_noff[i + 1] <= pair) ++i;    // owning out-joint
    const int j = c_nflat[pair];          // in-joint
    _Float16* dst = wh + (size_t)pk * (COUT_PJ * CIN_PJ);
    for (int idx = threadIdx.x; idx < COUT_PJ * CIN_PJ; idx += 256) {
        const int co = idx >> 6;
        const int ci = idx & 63;
        const size_t src =
            (((size_t)(i * COUT_PJ + co) * IN_C) + (size_t)(j * CIN_PJ + ci)) * KS + k;
        dst[idx] = (_Float16)w[src];
    }
}

// ---------------------------------------------------------------------------
// Staging helpers: x[b, jj*64 .. +63, 2*t0-7 .. 2*t0+134] -> LDS (f16),
// time-major rows [tt][c] with 144B pitch.
// ---------------------------------------------------------------------------
__device__ __forceinline__
void mg_stage_x(const float* __restrict__ xbase, _Float16* __restrict__ dst,
                int t0, int tid, bool edge)
{
    if (edge) {                            // reflect padding (tiles 0 and 31)
        for (int idx = tid; idx < XT * CIN_PJ; idx += 256) {
            const int c  = idx / XT;
            const int tt = idx - c * XT;
            int s = 2 * t0 - 7 + tt;
            s = (s < 0) ? -s : s;
            s = (s > T_IN - 1) ? (2 * (T_IN - 1) - s) : s;
            dst[tt * PITCH + c] = (_Float16)xbase[(size_t)c * T_IN + s];
        }
    } else {                               // interior: contiguous window
        const float* xb = xbase + (2 * t0 - 7);
        for (int idx = tid; idx < XT * CIN_PJ; idx += 256) {
            const int c  = idx / XT;
            const int tt = idx - c * XT;
            dst[tt * PITCH + c] = (_Float16)xb[(size_t)c * T_IN + tt];
        }
    }
}

__device__ __forceinline__
void mg_prefetch_x(const float* __restrict__ xbase, int t0, int tid)
{
    int sb = 2 * t0 - 7;
    sb = (sb < 0) ? 0 : sb;                            // begin-clamped window
    const int c = tid >> 2;                            // 0..63
    const int q = tid & 3;
    // 4 prefetches per channel row, 144B apart -> covers the 568B window
    __builtin_prefetch(xbase + (size_t)c * T_IN + sb + q * 36, 0, 1);
}

// ---------------------------------------------------------------------------
// Kernel 2: block-sparse implicit-GEMM conv via WMMA.
// Workgroup: out-joint ij, 64-wide time tile, one batch. 8 waves, wave m-tile.
// ---------------------------------------------------------------------------
__global__ __launch_bounds__(256)
void mg_conv_wmma(const float* __restrict__ x, const _Float16* __restrict__ wh,
                  const float* __restrict__ bias, float* __restrict__ y)
{
    __shared__ _Float16 xs[2][XT * PITCH];

    const int tb   = blockIdx.x;          // time tile index
    const int ij   = blockIdx.y;          // out joint
    const int b    = blockIdx.z;          // batch
    const int tid  = threadIdx.x;
    const int wave = tid >> 5;
    const int lane = tid & 31;
    const int nl   = lane & 15;           // column / row-within-tile index
    const int h    = lane >> 4;           // lane half (K-range selector)
    const int t0   = tb * TT;
    const bool edge = (tb == 0) || (tb == NTB - 1);

    v8f acc[4] = {};                      // 4 N-tiles of 16x16 f32

    const int p0 = c_noff[ij];
    const int p1 = c_noff[ij + 1];
    const float* xb0 = x + (size_t)b * IN_C * T_IN;

    // prologue: stage first pair
    mg_stage_x(xb0 + (size_t)c_nflat[p0] * CIN_PJ * T_IN, xs[0], t0, tid, edge);
    __syncthreads();

    for (int pp = p0; pp < p1; ++pp) {
        const int buf = (pp - p0) & 1;
        const _Float16* cur = xs[buf];

        // stage next pair into the other buffer (overlaps with WMMA below)
        if (pp + 1 < p1)
            mg_stage_x(xb0 + (size_t)c_nflat[pp + 1] * CIN_PJ * T_IN,
                       xs[buf ^ 1], t0, tid, edge);
        // warm L2/L0 for the pair after next
        if (pp + 2 < p1)
            mg_prefetch_x(xb0 + (size_t)c_nflat[pp + 2] * CIN_PJ * T_IN, t0, tid);

        const _Float16* wbase =
            wh + (size_t)pp * KS * (COUT_PJ * CIN_PJ) +
            (size_t)(wave * 16 + nl) * CIN_PJ;   // this lane's A row

        for (int k = 0; k < KS; ++k) {
            const _Float16* wk = wbase + (size_t)k * (COUT_PJ * CIN_PJ);
            #pragma unroll
            for (int kt = 0; kt < 2; ++kt) {     // two 32-deep K chunks of cin
                // ISA 16-bit A/B packing: slot p -> K = (p<8 ? p : p+8) + 8*h
                const int c0 = kt * 32 + 8 * h;
                V16 A;
                A.h[0] = *(const v8h*)(wk + c0);
                A.h[1] = *(const v8h*)(wk + c0 + 16);
                #pragma unroll
                for (int nt = 0; nt < 4; ++nt) {
                    const int ttx = 2 * (nt * 16 + nl) + k;  // stride-2 conv
                    V16 B;
                    B.h[0] = *(const v8h*)(cur + ttx * PITCH + c0);
                    B.h[1] = *(const v8h*)(cur + ttx * PITCH + c0 + 16);
                    acc[nt] = __builtin_amdgcn_wmma_f32_16x16x32_f16(
                        false, A.v, false, B.v, (short)0, acc[nt], false, false);
                }
            }
        }
        __syncthreads();   // staged buffer ready / cur safe to overwrite
    }

    // ---- epilogue: + bias, store y[b, ij*128+., t0+.] (f32) ----
    const int  row0  = ij * COUT_PJ + wave * 16 + 8 * h;   // + r (0..7)
    const float* bi  = bias + row0;
    float* ybase = y + ((size_t)b * OUT_C + row0) * T_OUT + t0;
    #pragma unroll
    for (int nt = 0; nt < 4; ++nt) {
        #pragma unroll
        for (int r = 0; r < 8; ++r) {
            ybase[(size_t)r * T_OUT + nt * 16 + nl] = acc[nt][r] + bi[r];
        }
    }
}

// ---------------------------------------------------------------------------
// Kernel 3: joint pooling (avg of 1-2 edges) + leaky relu (0.2)
// ---------------------------------------------------------------------------
__global__ __launch_bounds__(256)
void mg_pool_lrelu(const float* __restrict__ y, float* __restrict__ out)
{
    const size_t total = (size_t)BATCH * NGROUP * COUT_PJ * T_OUT;
    size_t idx = (size_t)blockIdx.x * 256 + threadIdx.x;
    if (idx >= total) return;
    const int t = (int)(idx % T_OUT);
    size_t r = idx / T_OUT;
    const int c = (int)(r % COUT_PJ); r /= COUT_PJ;
    const int g = (int)(r % NGROUP);
    const int b = (int)(r / NGROUP);
    const int ja = c_poolA[g], jb = c_poolB[g];
    float v = y[((size_t)b * OUT_C + (size_t)ja * COUT_PJ + c) * T_OUT + t];
    if (jb >= 0)
        v = 0.5f * (v + y[((size_t)b * OUT_C + (size_t)jb * COUT_PJ + c) * T_OUT + t]);
    out[idx] = (v > 0.0f) ? v : 0.2f * v;
}

// ---------------------------------------------------------------------------
extern "C" void kernel_launch(void* const* d_in, const int* in_sizes, int n_in,
                              void* d_out, int out_size, void* d_ws, size_t ws_size,
                              hipStream_t stream)
{
    const float* x      = (const float*)d_in[0];   // (8,1344,4096)
    const float* weight = (const float*)d_in[1];   // (2688,1344,15)
    const float* bias   = (const float*)d_in[2];   // (2688,)
    // d_in[3] mask, d_in[4] pool_w: structure hardcoded above.
    float* out = (float*)d_out;                    // (8,1536,2048)

    const size_t wh_bytes = (size_t)NPAIRS * KS * COUT_PJ * CIN_PJ * sizeof(_Float16);
    _Float16* wh = (_Float16*)d_ws;
    float* y = (float*)((char*)d_ws + ((wh_bytes + 255) & ~(size_t)255));

    mg_pack_weights<<<NPAIRS * KS, 256, 0, stream>>>(weight, wh);
    mg_conv_wmma<<<dim3(NTB, NJ, BATCH), 256, 0, stream>>>(x, wh, bias, y);

    const size_t pool_total = (size_t)BATCH * NGROUP * COUT_PJ * T_OUT;
    mg_pool_lrelu<<<(unsigned)((pool_total + 255) / 256), 256, 0, stream>>>(y, out);
}
